// Kernel_4285503710782699712_53472342835838
// MI455X (gfx1250) — matrix operand order repaired and checked by reading, not compiled, not run on hardware
//
#include <hip/hip_runtime.h>

// ---------------------------------------------------------------------------
// MI455X (gfx1250) implementation.
//
// Roofline: ~29 GFLOP vs 51 MB unavoidable HBM traffic (2.2 us @ 23.3 TB/s)
// => compute bound on the matrix pipes. The two dominant GEMMs (t3 corr GEMM
// K=3136, t10 GEMM K=896) use bf16 split-precision (x = hi + lo, compute
// hi*hi + hi*lo + lo*hi, fp32 accumulate) on V_WMMA_F32_16X16X32_BF16 -- 8x K
// per instruction vs the f32 WMMA, net ~2.7x fewer matrix-pipe cycles at
// ~fp32 accuracy. The small t2 conv (17% of FLOPs) stays on
// V_WMMA_F32_16X16X4_F32. All intermediates stay in the 192MB L2 via d_ws.
// ---------------------------------------------------------------------------

typedef float  v2f   __attribute__((ext_vector_type(2)));
typedef float  v8f   __attribute__((ext_vector_type(8)));
typedef __bf16 v16bf __attribute__((ext_vector_type(16)));

#define NB 16
#define CC 128
#define HH 56
#define WW 56
#define HP 62   // t2 padded height: 3 zero rows + 56 + 3 zero rows

// workspace layout in floats
#define T2P_OFF 0
#define T2P_SZ  (NB*CC*HP*WW)              // 7,110,656 floats
#define T3_OFF  (T2P_OFF + T2P_SZ + 64)    // +64 pad for partial-tile tail reads
#define T3_SZ   (NB*7*CC*CC)               // 1,835,008
#define T9N     (NB*CC*896)                // t9T element count (bf16 each)
#define T9H_OFF (T3_OFF + T3_SZ)           // bf16 hi array (T9N/2 floats)
#define T9L_OFF (T9H_OFF + T9N/2)          // bf16 lo array (T9N/2 floats)

static __device__ inline v8f wmma4(v2f a, v2f b, v8f c) {
  return __builtin_amdgcn_wmma_f32_16x16x4_f32(false, a, false, b, (short)0, c,
                                               false, false);
}
static __device__ inline v8f wmma32(v16bf a, v16bf b, v8f c) {
  return __builtin_amdgcn_wmma_f32_16x16x32_bf16(false, a, false, b, (short)0, c,
                                                 false, false);
}
static __device__ inline v8f vzero8() {
  v8f v = {0.f, 0.f, 0.f, 0.f, 0.f, 0.f, 0.f, 0.f};
  return v;
}

// bf16 split helpers (explicit RNE bit math; __bf16 used only as WMMA carrier)
static __device__ inline unsigned bf16hi(float f) {
  union { float f; unsigned u; } v; v.f = f;
  unsigned u = v.u;
  return (u + 0x7fffu + ((u >> 16) & 1u)) >> 16;
}
static __device__ inline float bf16tof(unsigned bits) {
  union { float f; unsigned u; } v; v.u = bits << 16; return v.f;
}
static __device__ inline void split2(float f, unsigned& h, unsigned& l) {
  h = bf16hi(f);
  l = bf16hi(f - bf16tof(h));
}

union BfPack { int i[8]; v16bf v; };  // 16 bf16 = 8 dwords

// ---------------------------------------------------------------------------
// Zero the height-pad rows of t2p (rows 0..2 and 59..61 per (n,c)) + tail pad.
// ---------------------------------------------------------------------------
__global__ void kZeroPad(float* __restrict__ ws) {
  int idx = blockIdx.x * 256 + threadIdx.x;
  const int total = NB * CC * 6 * WW;
  if (idx < total) {
    int w  = idx % WW;
    int t  = idx / WW;
    int rr = t % 6;
    int nc = t / 6;
    int row = (rr < 3) ? rr : (rr + 56);   // 0,1,2,59,60,61
    ws[T2P_OFF + ((size_t)nc * HP + row) * WW + w] = 0.f;
  } else if (idx < total + 64) {
    ws[T2P_OFF + T2P_SZ + (idx - total)] = 0.f;
  }
}

// ---------------------------------------------------------------------------
// t2 = dense (1,3) conv, padding (0,1). One WG per (n,h). fp32 WMMA.
// GEMM: C[o,w] += sum_{K=c*3+dx} w2[o,K] * xrow[c, w+dx-1]
// ---------------------------------------------------------------------------
__global__ void __launch_bounds__(256) kConv13(const float* __restrict__ x,
                                               const float* __restrict__ w2,
                                               float* __restrict__ ws) {
  __shared__ float sx[CC * 72];  // pitch 72: [0..3]=0, [4..59]=x, [60..71]=0
  const int n = blockIdx.x / HH, h = blockIdx.x % HH;
  const int tid = threadIdx.x;
  const float* xrow = x + ((size_t)n * CC * HH + h) * WW;
  for (int i = tid; i < CC * 72; i += 256) {  // 36 iters, wave-uniform
    int c = i / 72, j = i % 72;
    sx[i] = (j >= 4 && j < 60) ? xrow[(size_t)c * HH * WW + (j - 4)] : 0.f;
  }
  __syncthreads();

  const int lane = tid & 31, wave = tid >> 5;
  const int ln = lane & 15, hi = lane >> 4;
  v8f acc[4];
  #pragma unroll
  for (int t = 0; t < 4; ++t) acc[t] = vzero8();

  const float* aRow = w2 + (size_t)(wave * 16 + ln) * 384;  // w2[o][c*3+dx]
  for (int ks = 0; ks < 96; ++ks) {
    const int K0 = ks * 4 + hi * 2;
    v2f A = *(const v2f*)(aRow + K0);
    int off[2];
    #pragma unroll
    for (int e = 0; e < 2; ++e) {
      int K  = K0 + e;
      int c  = (int)((unsigned)K / 3u);
      int dx = K - 3 * c;
      off[e] = c * 72 + 3 + dx;
    }
    #pragma unroll
    for (int t = 0; t < 4; ++t) {
      int w0 = t * 16 + ln;
      v2f B = {sx[off[0] + w0], sx[off[1] + w0]};
      acc[t] = wmma4(A, B, acc[t]);
    }
  }

  float* t2p = ws + T2P_OFF;
  #pragma unroll
  for (int t = 0; t < 4; ++t) {
    int w = t * 16 + ln;
    if (w < WW) {
      #pragma unroll
      for (int r = 0; r < 8; ++r) {
        int o = wave * 16 + r + hi * 8;
        t2p[(((size_t)n * CC + o) * HP + (h + 3)) * WW + w] = acc[t][r];
      }
    }
  }
}

// ---------------------------------------------------------------------------
// t3[n,a,k,b] = sum_{h,w} x[n,a,h,w+2k-6] * x[n,b,h,w]   (zero padded width)
// One WG per (n,k). Row staged in LDS as packed bf16 hi/lo pairs with zero
// pads covering both the +-6 shift and K padding 56->64 (2 K-steps of 32 per
// row). Each wave owns a 2x4 tile block; 24 bf16x3 WMMAs per K-step.
// LDS bf16 pitch = 80 (40 ints): idx [0..5]=0, [6..61]=x row, [62..79]=0.
// ---------------------------------------------------------------------------
__global__ void __launch_bounds__(256) kCorr(const float* __restrict__ x,
                                             float* __restrict__ ws) {
  __shared__ int sH[CC * 40];
  __shared__ int sL[CC * 40];
  const int n = blockIdx.x / 7, kt = blockIdx.x % 7;
  const int d = 2 * kt - 6;
  const int tid = threadIdx.x;
  const int lane = tid & 31, wave = tid >> 5;
  const int ln = lane & 15, hi = lane >> 4;
  const int ti = wave >> 1;  // a-tiles 2ti, 2ti+1
  const int tj = wave & 1;   // b-tiles 4tj .. 4tj+3

  v8f acc[2][4];
  #pragma unroll
  for (int s = 0; s < 2; ++s)
    #pragma unroll
    for (int t = 0; t < 4; ++t) acc[s][t] = vzero8();

  const float* xn = x + (size_t)n * CC * HH * WW;
  const int aRow0 = (2 * ti) * 16 * 40 + ln * 40;
  const int aRow1 = aRow0 + 16 * 40;
  const int bRow0 = (4 * tj) * 16 * 40 + ln * 40;

  for (int h = 0; h < HH; ++h) {
    __syncthreads();
    for (int i = tid; i < CC * 40; i += 256) {  // 20 iters, wave-uniform
      int c = i / 40, q = i % 40;
      int j0 = 2 * q;
      const float* xr = xn + ((size_t)c * HH + h) * WW;
      float f0 = (j0 >= 6 && j0 < 62) ? xr[j0 - 6] : 0.f;
      float f1 = (j0 >= 6 && j0 <= 60) ? xr[j0 - 5] : 0.f;  // j0+1 in [6,62)
      unsigned h0, l0, h1, l1;
      split2(f0, h0, l0);
      split2(f1, h1, l1);
      sH[i] = (int)(h0 | (h1 << 16));
      sL[i] = (int)(l0 | (l1 << 16));
    }
    __syncthreads();

    #pragma unroll
    for (int kq = 0; kq < 2; ++kq) {
      const int k0 = kq * 32;
      // A: two 8-K runs per lane: bf16 idx = 6+d+k0+hi*8 (+16); always even.
      const int aBase = (6 + d + k0 + hi * 8) >> 1;  // int index
      BfPack A0h, A0l, A1h, A1l;
      #pragma unroll
      for (int q = 0; q < 4; ++q) {
        A0h.i[q]     = sH[aRow0 + aBase + q];
        A0h.i[4 + q] = sH[aRow0 + aBase + 8 + q];
        A0l.i[q]     = sL[aRow0 + aBase + q];
        A0l.i[4 + q] = sL[aRow0 + aBase + 8 + q];
        A1h.i[q]     = sH[aRow1 + aBase + q];
        A1h.i[4 + q] = sH[aRow1 + aBase + 8 + q];
        A1l.i[q]     = sL[aRow1 + aBase + q];
        A1l.i[4 + q] = sL[aRow1 + aBase + 8 + q];
      }
      // B: two 8-K runs per lane: bf16 idx = 6+k0+hi*8 (+16) (even).
      const int bBase = ((6 + k0) >> 1) + hi * 4;
      #pragma unroll
      for (int t = 0; t < 4; ++t) {
        BfPack Bh, Bl;
        #pragma unroll
        for (int q = 0; q < 8; ++q) {
          Bh.i[q] = sH[bRow0 + t * 16 * 40 + bBase + q + (q >> 2) * 4];
          Bl.i[q] = sL[bRow0 + t * 16 * 40 + bBase + q + (q >> 2) * 4];
        }
        // interleave the two accumulator rows to space same-dest RAW hazards
        acc[0][t] = wmma32(A0h.v, Bh.v, acc[0][t]);
        acc[1][t] = wmma32(A1h.v, Bh.v, acc[1][t]);
        acc[0][t] = wmma32(A0h.v, Bl.v, acc[0][t]);
        acc[1][t] = wmma32(A1h.v, Bl.v, acc[1][t]);
        acc[0][t] = wmma32(A0l.v, Bh.v, acc[0][t]);
        acc[1][t] = wmma32(A1l.v, Bh.v, acc[1][t]);
      }
    }
  }

  float* t3 = ws + T3_OFF;  // [n][k][a][b]
  #pragma unroll
  for (int s = 0; s < 2; ++s)
    #pragma unroll
    for (int t = 0; t < 4; ++t)
      #pragma unroll
      for (int r = 0; r < 8; ++r) {
        int a = (2 * ti + s) * 16 + r + hi * 8;
        int b = (4 * tj + t) * 16 + ln;
        t3[(((size_t)n * 7 + kt) * CC + a) * CC + b] = acc[s][t][r];
      }
}

// ---------------------------------------------------------------------------
// t9T[n][b][a*7+k] = t3[n][k][a][b]*p6w[a,k]*p9w[k,b]/(sqrt(3136)*sqrt(896)),
// written as bf16 hi/lo arrays so kMain's A operand is pure packed loads.
// ---------------------------------------------------------------------------
__global__ void kScale(const float* __restrict__ p6w,
                       const float* __restrict__ p9w,
                       float* __restrict__ ws) {
  int idx = blockIdx.x * 256 + threadIdx.x;
  if (idx >= T9N) return;
  int K = idx % 896;
  int t = idx / 896;
  int b = t % CC;
  int n = t / CC;
  int a = (int)((unsigned)K / 7u);
  int k = K - 7 * a;
  const float S = 1.0f / (56.0f * 29.933259094191532f);  // 1/(sqrt(HW)*sqrt(7C))
  float v = ws[T3_OFF + (((size_t)n * 7 + k) * CC + a) * CC + b]
          * p6w[a * 7 + k] * p9w[k * CC + b] * S;
  unsigned hb, lb;
  split2(v, hb, lb);
  ((unsigned short*)(ws + T9H_OFF))[idx] = (unsigned short)hb;
  ((unsigned short*)(ws + T9L_OFF))[idx] = (unsigned short)lb;
}

// ---------------------------------------------------------------------------
// t10[n,b,h,w] = sum_{a,k} t9T[n,b,a*7+k] * t2[n,a,h+k-3,w]  (padded t2 rows)
// A from bf16 t9T (packed, 16B-aligned); B gathered fp32 from t2p and split
// to bf16 in-register; 3 WMMAs per tile per K-step of 32 (28 K-steps).
// Fused epilogue: out = t8 - t10 + t7 (depthwise convs on VALU).
// ---------------------------------------------------------------------------
__global__ void __launch_bounds__(256) kMain(const float* __restrict__ x,
                                             const float* __restrict__ w7,
                                             const float* __restrict__ w8,
                                             const float* __restrict__ ws,
                                             float* __restrict__ out) {
  const int n = blockIdx.x / HH, h = blockIdx.x % HH;
  const int tid = threadIdx.x;
  const int lane = tid & 31, wave = tid >> 5;
  const int ln = lane & 15, hi = lane >> 4;
  const float* t2n = ws + T2P_OFF + (size_t)n * CC * HP * WW;

  v8f acc[4];
  #pragma unroll
  for (int t = 0; t < 4; ++t) acc[t] = vzero8();

  const unsigned short* t9h = (const unsigned short*)(ws + T9H_OFF);
  const unsigned short* t9l = (const unsigned short*)(ws + T9L_OFF);
  const int bRowA = n * CC + wave * 16 + ln;
  const int* aH = (const int*)(t9h + (size_t)bRowA * 896);
  const int* aL = (const int*)(t9l + (size_t)bRowA * 896);

  for (int ks = 0; ks < 28; ++ks) {
    const int K0 = ks * 32;
    // A: two 8-K runs per lane at bf16 idx K0+hi*8 (+16); int idx mult of 4.
    const int ab = (K0 >> 1) + hi * 4;
    BfPack Ah, Al;
    #pragma unroll
    for (int q = 0; q < 4; ++q) {
      Ah.i[q]     = aH[ab + q];
      Ah.i[4 + q] = aH[ab + 8 + q];
      Al.i[q]     = aL[ab + q];
      Al.i[4 + q] = aL[ab + 8 + q];
    }
    // B row offsets for this lane-half's 16 K values (K=(a,k) -> t2 row h+k)
    const int Kb = K0 + hi * 8;
    int rowO[16];
    #pragma unroll
    for (int j = 0; j < 16; ++j) {
      int K  = Kb + j + (j >> 3) * 8;
      int a  = (int)((unsigned)K / 7u);
      int kj = K - 7 * a;
      rowO[j] = (a * HP + (h + kj)) * WW;
    }
    #pragma unroll
    for (int t = 0; t < 4; ++t) {
      const int w0 = t * 16 + ln;
      BfPack Bh, Bl;
      #pragma unroll
      for (int q = 0; q < 8; ++q) {
        float f0 = t2n[rowO[2 * q] + w0];
        float f1 = t2n[rowO[2 * q + 1] + w0];
        unsigned h0, l0, h1, l1;
        split2(f0, h0, l0);
        split2(f1, h1, l1);
        Bh.i[q] = (int)(h0 | (h1 << 16));
        Bl.i[q] = (int)(l0 | (l1 << 16));
      }
      acc[t] = wmma32(Ah.v, Bh.v, acc[t]);
      acc[t] = wmma32(Ah.v, Bl.v, acc[t]);
      acc[t] = wmma32(Al.v, Bh.v, acc[t]);
    }
  }

  // ---- fused epilogue: out = t8 - t10 + t7 ----
  const float* xn = x + (size_t)n * CC * HH * WW;
  #pragma unroll
  for (int r = 0; r < 8; ++r) {
    int b = wave * 16 + r + hi * 8;
    float w7v[5];
    #pragma unroll
    for (int j = 0; j < 5; ++j) w7v[j] = w7[b * 5 + j];
    float w8v[25];
    #pragma unroll
    for (int j = 0; j < 25; ++j) w8v[j] = w8[b * 25 + j];
    const float* xb  = xn + (size_t)b * HH * WW;
    const float* t2b = t2n + (size_t)b * HP * WW + (size_t)(h + 3) * WW;
    #pragma unroll
    for (int t = 0; t < 4; ++t) {
      int w = t * 16 + ln;
      if (w >= WW) continue;
      float t7 = 0.f;
      #pragma unroll
      for (int j = 0; j < 5; ++j) {
        int wc = w + j - 2;
        if ((unsigned)wc < (unsigned)WW) t7 += t2b[wc] * w7v[j];
      }
      float t8 = 0.f;
      #pragma unroll
      for (int i = 0; i < 5; ++i) {
        int hc = h + 2 * i - 4;
        if ((unsigned)hc < (unsigned)HH) {
          #pragma unroll
          for (int j = 0; j < 5; ++j) {
            int wc = w + 2 * j - 4;
            if ((unsigned)wc < (unsigned)WW)
              t8 += xb[(size_t)hc * WW + wc] * w8v[i * 5 + j];
          }
        }
      }
      out[(((size_t)n * CC + b) * HH + h) * WW + w] = t8 - acc[t][r] + t7;
    }
  }
}

// ---------------------------------------------------------------------------
extern "C" void kernel_launch(void* const* d_in, const int* in_sizes, int n_in,
                              void* d_out, int out_size, void* d_ws, size_t ws_size,
                              hipStream_t stream) {
  (void)in_sizes; (void)n_in; (void)out_size; (void)ws_size;
  const float* x   = (const float*)d_in[0];
  const float* w2  = (const float*)d_in[1];
  const float* p6w = (const float*)d_in[2];
  const float* w7  = (const float*)d_in[3];
  const float* w8  = (const float*)d_in[4];
  const float* p9w = (const float*)d_in[5];
  float* out = (float*)d_out;
  float* ws  = (float*)d_ws;

  kZeroPad<<<dim3((NB * CC * 6 * WW + 64 + 255) / 256), dim3(256), 0, stream>>>(ws);
  kConv13<<<dim3(NB * HH), dim3(256), 0, stream>>>(x, w2, ws);
  kCorr<<<dim3(NB * 7), dim3(256), 0, stream>>>(x, ws);
  kScale<<<dim3((T9N + 255) / 256), dim3(256), 0, stream>>>(p6w, p9w, ws);
  kMain<<<dim3(NB * HH), dim3(256), 0, stream>>>(x, w7, w8, ws, out);
}
